// TransformerBlock_2765958939264
// MI455X (gfx1250) — matrix operand order repaired and checked by reading, not compiled, not run on hardware
//
#include <hip/hip_runtime.h>
#include <hip/hip_bf16.h>

// ---------------------------------------------------------------------------
// MI455X (gfx1250) implementation of the DCT-domain Restormer block.
//  - All GEMMs (DCT/IDCT, conv1x1s, window attention) run on
//    v_wmma_f32_16x16x32_f16 (f16 operands, f32 accumulate).
//  - GEMM uses a double-buffered LDS pipeline fed by
//    GLOBAL_LOAD_ASYNC_TO_LDS_B128 (ASYNCcnt) when the builtin exists,
//    falling back to global_load+ds_store otherwise.
//  - Memory-bound workload (~0.9e11 FLOP vs ~2.5 GB moved @23.3TB/s), so f16
//    operand compression + buffer aliasing matter more than peak math.
// ---------------------------------------------------------------------------

typedef __attribute__((ext_vector_type(16))) _Float16     v16h;
typedef __attribute__((ext_vector_type(8)))  float        v8f;
typedef __attribute__((ext_vector_type(4)))  unsigned int u32x4;
typedef __attribute__((ext_vector_type(4)))  float        f32x4;

union Frag16 { v16h v; u32x4 u[2]; };
union H8 { u32x4 u; _Float16 h[8]; };

__device__ inline v16h frag_ld(const _Float16* p0, const _Float16* p1) {
  Frag16 f;
  f.u[0] = *(const u32x4*)p0;
  f.u[1] = *(const u32x4*)p1;
  return f.v;
}

#if defined(__has_builtin)
#if __has_builtin(__builtin_amdgcn_global_load_async_to_lds_b128) && \
    __has_builtin(__builtin_amdgcn_s_wait_asynccnt)
#define HAVE_ASYNC 1
#endif
#endif
#ifndef HAVE_ASYNC
#define HAVE_ASYNC 0
#endif

#if HAVE_ASYNC
// Builtin prototype (from hipcc diagnostic): (v4i AS(1)*, v4i AS(3)*, Ii, Ii)
typedef int v4i __attribute__((vector_size(16)));
typedef __attribute__((address_space(1))) void gas_void;
typedef __attribute__((address_space(3))) void las_void;
typedef __attribute__((address_space(1))) v4i* gptr_v4i;
typedef __attribute__((address_space(3))) v4i* lptr_v4i;
__device__ inline void async_g2l_b128(const _Float16* g, _Float16* l) {
  __builtin_amdgcn_global_load_async_to_lds_b128(
      (gptr_v4i)(gas_void*)(void*)g,
      (lptr_v4i)(las_void*)(void*)l, 0, 0);
}
#endif

#define LDP 40  // LDS row pitch (f16 elems): 80B = 5*16B, keeps 16B alignment

// Generic batched GEMM: D[M,N] = A[M,K] x B + bias, A row-major [M,K].
// bNK=1: B given as [N,K] row-major; bNK=0: B given as [K,N] row-major.
// Block tile 64x64, 4 waves, each wave 32x32 (four 16x16 WMMA accumulators).
// Double-buffered LDS software pipeline.
__global__ __launch_bounds__(128)
void gemm16(const _Float16* __restrict__ A, const _Float16* __restrict__ B,
            float* __restrict__ Df, _Float16* __restrict__ Dh,
            const float* __restrict__ bias,
            int Mreal, int N, int K,
            long long sA, long long sB, long long sD, int bNK)
{
  __shared__ alignas(16) _Float16 lsA[2][64 * LDP];
  __shared__ alignas(16) _Float16 lsB[2][64 * LDP];

  const int tid  = threadIdx.x;
  const int lane = tid & 31;
  const int wid  = tid >> 5;

  A += (long long)blockIdx.z * sA;
  B += (long long)blockIdx.z * sB;

  const int m0 = blockIdx.y * 64;
  const int n0 = blockIdx.x * 64;
  const int wm = (wid & 1) << 5;   // 0,32
  const int wn = (wid >> 1) << 5;  // 0,32
  const int r16 = lane & 15;
  const int kh  = lane >> 4;       // 0/1

  // cooperative-load thread mapping (128 threads)
  const int ra  = tid & 63;          // row (A tile / B^T tile)
  const int ks  = (tid >> 6) << 4;   // k segment: 0 / 16
  const int kb  = tid & 31;          // k (B [K,N] tile)
  const int nsg = (tid >> 5) << 4;   // n segment: 0,16,32,48

  v8f acc[2][2];
  const v8f z8 = {0.f,0.f,0.f,0.f,0.f,0.f,0.f,0.f};
  acc[0][0] = z8; acc[0][1] = z8; acc[1][0] = z8; acc[1][1] = z8;

  const int nk = K >> 5;

  u32x4 arg[2][2];   // A staging regs (fallback path)
  u32x4 brg[2][2];   // B staging regs (B [K,N] transpose path)

  // Issue tile (kk) into LDS buffer pb (async) / staging regs (sync).
  auto issue = [&](int kk, int pb) {
#if HAVE_ASYNC
    async_g2l_b128(A + (long long)(m0 + ra) * K + kk + ks,
                   &lsA[pb][ra * LDP + ks]);
    async_g2l_b128(A + (long long)(m0 + ra) * K + kk + ks + 8,
                   &lsA[pb][ra * LDP + ks + 8]);
    if (bNK) {
      async_g2l_b128(B + (long long)(n0 + ra) * K + kk + ks,
                     &lsB[pb][ra * LDP + ks]);
      async_g2l_b128(B + (long long)(n0 + ra) * K + kk + ks + 8,
                     &lsB[pb][ra * LDP + ks + 8]);
    } else {
      brg[pb][0] = *(const u32x4*)(B + (long long)(kk + kb) * N + n0 + nsg);
      brg[pb][1] = *(const u32x4*)(B + (long long)(kk + kb) * N + n0 + nsg + 8);
      if (kk + 64 < K)  // stream-prefetch two tiles ahead
        __builtin_prefetch(B + (long long)(kk + 64 + kb) * N + n0 + nsg, 0, 1);
    }
#else
    arg[pb][0] = *(const u32x4*)(A + (long long)(m0 + ra) * K + kk + ks);
    arg[pb][1] = *(const u32x4*)(A + (long long)(m0 + ra) * K + kk + ks + 8);
    if (bNK) {
      brg[pb][0] = *(const u32x4*)(B + (long long)(n0 + ra) * K + kk + ks);
      brg[pb][1] = *(const u32x4*)(B + (long long)(n0 + ra) * K + kk + ks + 8);
    } else {
      brg[pb][0] = *(const u32x4*)(B + (long long)(kk + kb) * N + n0 + nsg);
      brg[pb][1] = *(const u32x4*)(B + (long long)(kk + kb) * N + n0 + nsg + 8);
      if (kk + 64 < K)
        __builtin_prefetch(B + (long long)(kk + 64 + kb) * N + n0 + nsg, 0, 1);
    }
#endif
  };

  // Commit tile in buffer pb: finish its data movement, then barrier.
  auto commit = [&](int pb, bool more) {
#if !HAVE_ASYNC
    *(u32x4*)&lsA[pb][ra * LDP + ks]     = arg[pb][0];
    *(u32x4*)&lsA[pb][ra * LDP + ks + 8] = arg[pb][1];
    if (bNK) {
      *(u32x4*)&lsB[pb][ra * LDP + ks]     = brg[pb][0];
      *(u32x4*)&lsB[pb][ra * LDP + ks + 8] = brg[pb][1];
    }
#endif
    if (!bNK) {  // transpose-scatter B [K,N] -> LDS [n][k]
      H8 t0, t1; t0.u = brg[pb][0]; t1.u = brg[pb][1];
#pragma unroll
      for (int j = 0; j < 8; ++j) {
        lsB[pb][(nsg + j) * LDP + kb]     = t0.h[j];
        lsB[pb][(nsg + 8 + j) * LDP + kb] = t1.h[j];
      }
    }
#if HAVE_ASYNC
    if (more) {
      if (bNK) __builtin_amdgcn_s_wait_asynccnt(4);
      else     __builtin_amdgcn_s_wait_asynccnt(2);
    } else {
      __builtin_amdgcn_s_wait_asynccnt(0);
    }
#else
    (void)more;
#endif
    __syncthreads();
  };

  issue(0, 0);
  for (int i = 0; i < nk; ++i) {
    const int pb = i & 1;
    const bool more = (i + 1) < nk;
    if (more) issue((i + 1) << 5, pb ^ 1);
    commit(pb, more);

    const _Float16* La = lsA[pb];
    const _Float16* Lb = lsB[pb];
    const v16h a0 = frag_ld(&La[(wm + r16) * LDP + kh * 8],
                            &La[(wm + r16) * LDP + 16 + kh * 8]);
    const v16h a1 = frag_ld(&La[(wm + 16 + r16) * LDP + kh * 8],
                            &La[(wm + 16 + r16) * LDP + 16 + kh * 8]);
    const v16h b0 = frag_ld(&Lb[(wn + r16) * LDP + kh * 8],
                            &Lb[(wn + r16) * LDP + 16 + kh * 8]);
    const v16h b1 = frag_ld(&Lb[(wn + 16 + r16) * LDP + kh * 8],
                            &Lb[(wn + 16 + r16) * LDP + 16 + kh * 8]);
    acc[0][0] = __builtin_amdgcn_wmma_f32_16x16x32_f16(false, a0, false, b0,
                                                       (short)0, acc[0][0], false, false);
    acc[0][1] = __builtin_amdgcn_wmma_f32_16x16x32_f16(false, a0, false, b1,
                                                       (short)0, acc[0][1], false, false);
    acc[1][0] = __builtin_amdgcn_wmma_f32_16x16x32_f16(false, a1, false, b0,
                                                       (short)0, acc[1][0], false, false);
    acc[1][1] = __builtin_amdgcn_wmma_f32_16x16x32_f16(false, a1, false, b1,
                                                       (short)0, acc[1][1], false, false);
    __syncthreads();  // protect LDS buffer reuse by tile i+2
  }

#pragma unroll
  for (int mi = 0; mi < 2; ++mi) {
#pragma unroll
    for (int r = 0; r < 8; ++r) {
      const int m = m0 + wm + mi * 16 + r + kh * 8;
      if (m < Mreal) {
        const int n = n0 + wn + r16;
        const float bv = bias ? bias[m] : 0.f;
        const long long o = (long long)blockIdx.z * sD + (long long)m * N + n;
        const float v0 = acc[mi][0][r] + bv;
        const float v1 = acc[mi][1][r] + bv;
        if (Df) { Df[o] = v0; Df[o + 16] = v1; }
        if (Dh) { Dh[o] = (_Float16)v0; Dh[o + 16] = (_Float16)v1; }
      }
    }
  }
}

// Window channel attention: one wave per (b, h1, w1, head).
// q,k l2-normalized over e=64; attn = (q k^T)*T[head]; softmax over d; out=attn v.
__global__ __launch_bounds__(32)
void winattn(const float* __restrict__ qkv, const float* __restrict__ temp,
             _Float16* __restrict__ out)
{
  int id = blockIdx.x;
  const int head = id % 6;  id /= 6;
  const int w1   = id % 24; id /= 24;
  const int h1   = id % 24;
  const int b    = id / 24;
  const int lane = threadIdx.x;
  const int r16  = lane & 15;
  const int kh   = lane >> 4;

  __shared__ alignas(16) _Float16 sQ[32 * 72];
  __shared__ alignas(16) _Float16 sK[32 * 72];
  __shared__ alignas(16) _Float16 sVt[64 * LDP];
  __shared__ float sS[32 * 32];
  __shared__ alignas(16) _Float16 sP[32 * LDP];

  const long long HW  = 36864;
  const long long spb = (long long)(h1 * 8) * 192 + w1 * 8;

  // Load q,k (normalized) and v (transposed). One channel row per lane.
#pragma unroll
  for (int z = 0; z < 3; ++z) {
    const long long base =
        ((long long)b * 576 + z * 192 + head * 32 + lane) * HW + spb;
    float buf[64];
#pragma unroll
    for (int i = 0; i < 8; ++i) {
      const f32x4 lo = *(const f32x4*)(qkv + base + i * 192);
      const f32x4 hi = *(const f32x4*)(qkv + base + i * 192 + 4);
      buf[i*8+0]=lo.x; buf[i*8+1]=lo.y; buf[i*8+2]=lo.z; buf[i*8+3]=lo.w;
      buf[i*8+4]=hi.x; buf[i*8+5]=hi.y; buf[i*8+6]=hi.z; buf[i*8+7]=hi.w;
    }
    if (z < 2) {
      float ss = 0.f;
#pragma unroll
      for (int e = 0; e < 64; ++e) ss += buf[e] * buf[e];
      const float inv = 1.f / fmaxf(sqrtf(ss), 1e-12f);
      _Float16* dst = (z == 0) ? sQ : sK;
#pragma unroll
      for (int e = 0; e < 64; ++e) dst[lane * 72 + e] = (_Float16)(buf[e] * inv);
    } else {
#pragma unroll
      for (int e = 0; e < 64; ++e) sVt[e * LDP + lane] = (_Float16)buf[e];
    }
  }
  __syncthreads();

  const v8f z8 = {0.f,0.f,0.f,0.f,0.f,0.f,0.f,0.f};
  v8f s00 = z8, s01 = z8, s10 = z8, s11 = z8;
#pragma unroll
  for (int kk = 0; kk < 64; kk += 32) {
    const v16h a0 = frag_ld(&sQ[r16 * 72 + kk + kh * 8],
                            &sQ[r16 * 72 + kk + 16 + kh * 8]);
    const v16h a1 = frag_ld(&sQ[(16 + r16) * 72 + kk + kh * 8],
                            &sQ[(16 + r16) * 72 + kk + 16 + kh * 8]);
    const v16h b0 = frag_ld(&sK[r16 * 72 + kk + kh * 8],
                            &sK[r16 * 72 + kk + 16 + kh * 8]);
    const v16h b1 = frag_ld(&sK[(16 + r16) * 72 + kk + kh * 8],
                            &sK[(16 + r16) * 72 + kk + 16 + kh * 8]);
    s00 = __builtin_amdgcn_wmma_f32_16x16x32_f16(false,a0,false,b0,(short)0,s00,false,false);
    s01 = __builtin_amdgcn_wmma_f32_16x16x32_f16(false,a0,false,b1,(short)0,s01,false,false);
    s10 = __builtin_amdgcn_wmma_f32_16x16x32_f16(false,a1,false,b0,(short)0,s10,false,false);
    s11 = __builtin_amdgcn_wmma_f32_16x16x32_f16(false,a1,false,b1,(short)0,s11,false,false);
  }

  const float ts = temp[head];
#pragma unroll
  for (int r = 0; r < 8; ++r) {
    const int m = r + kh * 8;
    sS[m * 32 + r16]             = s00[r] * ts;
    sS[m * 32 + 16 + r16]        = s01[r] * ts;
    sS[(m + 16) * 32 + r16]      = s10[r] * ts;
    sS[(m + 16) * 32 + 16 + r16] = s11[r] * ts;
  }
  __syncthreads();

  {  // softmax: one row per lane
    float tmp[32];
    float mx = -3.4e38f;
#pragma unroll
    for (int d = 0; d < 32; ++d) { tmp[d] = sS[lane * 32 + d]; mx = fmaxf(mx, tmp[d]); }
    float sum = 0.f;
#pragma unroll
    for (int d = 0; d < 32; ++d) { tmp[d] = __expf(tmp[d] - mx); sum += tmp[d]; }
    const float inv = 1.f / sum;
#pragma unroll
    for (int d = 0; d < 32; ++d) sP[lane * LDP + d] = (_Float16)(tmp[d] * inv);
  }
  __syncthreads();

  // out = P(32x32) @ V(32x64), K=32 -> one WMMA step, 2x4 fragments
  v8f o[2][4];
#pragma unroll
  for (int mt = 0; mt < 2; ++mt)
#pragma unroll
    for (int nt = 0; nt < 4; ++nt) o[mt][nt] = z8;

  const v16h pa0 = frag_ld(&sP[r16 * LDP + kh * 8], &sP[r16 * LDP + 16 + kh * 8]);
  const v16h pa1 = frag_ld(&sP[(16 + r16) * LDP + kh * 8],
                           &sP[(16 + r16) * LDP + 16 + kh * 8]);
#pragma unroll
  for (int nt = 0; nt < 4; ++nt) {
    const v16h bv = frag_ld(&sVt[(nt * 16 + r16) * LDP + kh * 8],
                            &sVt[(nt * 16 + r16) * LDP + 16 + kh * 8]);
    o[0][nt] = __builtin_amdgcn_wmma_f32_16x16x32_f16(false,pa0,false,bv,(short)0,o[0][nt],false,false);
    o[1][nt] = __builtin_amdgcn_wmma_f32_16x16x32_f16(false,pa1,false,bv,(short)0,o[1][nt],false,false);
  }

  const long long ob = ((long long)b * 192 + head * 32) * HW + spb;
#pragma unroll
  for (int mt = 0; mt < 2; ++mt)
#pragma unroll
    for (int nt = 0; nt < 4; ++nt)
#pragma unroll
      for (int r = 0; r < 8; ++r) {
        const int m = mt * 16 + r + kh * 8;
        const int e = nt * 16 + r16;
        out[ob + (long long)m * HW + (e >> 3) * 192 + (e & 7)] =
            (_Float16)o[mt][nt][r];
      }
}

// LayerNorm over channel dim of [2,192,192,192]; writes f16.
__global__ void ln_c(const float* __restrict__ x, const float* __restrict__ g,
                     const float* __restrict__ be, _Float16* __restrict__ y)
{
  const long long idx = (long long)blockIdx.x * blockDim.x + threadIdx.x;
  if (idx >= 2LL * 36864) return;
  const long long b = idx / 36864, hw = idx % 36864;
  const float* p = x + b * 192LL * 36864 + hw;
  float s = 0.f, s2 = 0.f;
  for (int c = 0; c < 192; ++c) {
    const float v = p[(long long)c * 36864];
    s += v; s2 += v * v;
  }
  const float mu  = s * (1.f / 192.f);
  const float var = s2 * (1.f / 192.f) - mu * mu;
  const float inv = rsqrtf(var + 1e-5f);
  _Float16* q = y + b * 192LL * 36864 + hw;
  for (int c = 0; c < 192; ++c)
    q[(long long)c * 36864] =
        (_Float16)((p[(long long)c * 36864] - mu) * inv * g[c] + be[c]);
}

// Depthwise 3x3, zero pad, channel-major f32.
__global__ void dwconv(const float* __restrict__ in, const float* __restrict__ w,
                       float* __restrict__ o, int Cc)
{
  const long long idx = (long long)blockIdx.x * blockDim.x + threadIdx.x;
  if (idx >= (long long)2 * Cc * 36864) return;
  const int hw = (int)(idx % 36864);
  const long long t = idx / 36864;
  const int c = (int)(t % Cc);
  const int h = hw / 192, ww = hw % 192;
  const float* wp = w + c * 9;
  const float* ip = in + t * 36864;
  float s = 0.f;
#pragma unroll
  for (int ki = 0; ki < 3; ++ki) {
    const int hh = h + ki - 1;
    if ((unsigned)hh < 192u) {
#pragma unroll
      for (int kj = 0; kj < 3; ++kj) {
        const int wc = ww + kj - 1;
        if ((unsigned)wc < 192u) s += wp[ki * 3 + kj] * ip[hh * 192 + wc];
      }
    }
  }
  o[idx] = s;
}

// Fused FFN: depthwise 3x3 on both gate halves + exact GELU gate -> f16,
// output padded to 512 channels (rows 510/511 zero) so K stays WMMA-friendly.
__global__ void dwgelu(const float* __restrict__ f, const float* __restrict__ w,
                       _Float16* __restrict__ g)
{
  const long long idx = (long long)blockIdx.x * blockDim.x + threadIdx.x;
  if (idx >= 2LL * 512 * 36864) return;
  const int hw = (int)(idx % 36864);
  const long long t = idx / 36864;
  const int c = (int)(t % 512);
  const int b = (int)(t / 512);
  if (c >= 510) { g[idx] = (_Float16)0.f; return; }
  const int h = hw / 192, ww = hw % 192;
  const float* w1 = w + c * 9;
  const float* w2 = w + (c + 510) * 9;
  const float* p1 = f + ((long long)b * 1020 + c) * 36864;
  const float* p2 = f + ((long long)b * 1020 + c + 510) * 36864;
  float s1 = 0.f, s2 = 0.f;
#pragma unroll
  for (int ki = 0; ki < 3; ++ki) {
    const int hh = h + ki - 1;
    if ((unsigned)hh < 192u) {
#pragma unroll
      for (int kj = 0; kj < 3; ++kj) {
        const int wc = ww + kj - 1;
        if ((unsigned)wc < 192u) {
          const int off = hh * 192 + wc;
          s1 += w1[ki * 3 + kj] * p1[off];
          s2 += w2[ki * 3 + kj] * p2[off];
        }
      }
    }
  }
  const float gl = 0.5f * s1 * (1.f + erff(s1 * 0.70710678118f));
  g[idx] = (_Float16)(gl * s2);
}

__global__ void addf(const float* __restrict__ a, const float* __restrict__ b,
                     float* __restrict__ o, long long n)
{
  const long long i = (long long)blockIdx.x * blockDim.x + threadIdx.x;
  if (i < n) o[i] = a[i] + b[i];
}

// f32 -> f16 with optional zero padding to (dr, dc).
__global__ void cvtpad(const float* __restrict__ s, _Float16* __restrict__ d,
                       int sr, int sc, int dr, int dc)
{
  const long long idx = (long long)blockIdx.x * blockDim.x + threadIdx.x;
  if (idx >= (long long)dr * dc) return;
  const int r = (int)(idx / dc), c = (int)(idx % dc);
  d[idx] = (r < sr && c < sc) ? (_Float16)s[r * sc + c] : (_Float16)0.f;
}

// Orthonormal DCT-II matrix (192x192) and its transpose, in f16.
__global__ void dctgen(_Float16* __restrict__ M, _Float16* __restrict__ MT)
{
  const int idx = blockIdx.x * blockDim.x + threadIdx.x;
  if (idx >= 192 * 192) return;
  const int i = idx / 192, j = idx % 192;
  float v;
  if (i == 0) v = sqrtf(1.f / 192.f);
  else v = cosf(3.14159265358979323846f * i * (2 * j + 1) / 384.f) *
           sqrtf(2.f / 192.f);
  M[i * 192 + j]  = (_Float16)v;
  MT[j * 192 + i] = (_Float16)v;
}

extern "C" void kernel_launch(void* const* d_in, const int* in_sizes, int n_in,
                              void* d_out, int out_size, void* d_ws, size_t ws_size,
                              hipStream_t stream)
{
  (void)in_sizes; (void)n_in; (void)out_size; (void)ws_size;

  const float* x     = (const float*)d_in[0];
  const float* n1w   = (const float*)d_in[1];
  const float* n1b   = (const float*)d_in[2];
  const float* wqkv  = (const float*)d_in[3];
  const float* wdw   = (const float*)d_in[4];
  const float* temp  = (const float*)d_in[5];
  const float* wproj = (const float*)d_in[6];
  const float* bproj = (const float*)d_in[7];
  const float* n2w   = (const float*)d_in[8];
  const float* n2b   = (const float*)d_in[9];
  const float* win   = (const float*)d_in[10];
  const float* wdwf  = (const float*)d_in[11];
  const float* wout  = (const float*)d_in[12];
  float* out = (float*)d_out;          // also hosts x1 (residual)
  char*  ws  = (char*)d_ws;

  const long long HW  = 36864;
  const long long NEL = 2LL * 192 * HW;     // 14,155,776

  // --- workspace arena (aliased; peak ~398 MiB) ---
  _Float16* M16     = (_Float16*)(ws + 0);
  _Float16* MT16    = (_Float16*)(ws + 73728);
  _Float16* wqkv16  = (_Float16*)(ws + 147456);
  _Float16* wproj16 = (_Float16*)(ws + 368640);
  _Float16* win16   = (_Float16*)(ws + 442368);   // padded to 1024x192
  _Float16* wout16  = (_Float16*)(ws + 835584);   // padded to 192x512
  const size_t WB  = 2u * 1024 * 1024;
  const size_t S28 = (size_t)NEL * 2;             // f16 activation slot
  _Float16* f16a = (_Float16*)(ws + WB);          // slot0
  _Float16* f16b = (_Float16*)(ws + WB + S28);    // slot1
  _Float16* gbuf = (_Float16*)(ws + WB);          // [2,512,HW] spans both slots
  const size_t WA = WB + (size_t)2 * 512 * HW * 2;
  float* qkv_pre = (float*)(ws + WA);
  float* qkvb    = (float*)(ws + WA + (size_t)2 * 576 * HW * 4);
  float* a4f     = (float*)(ws + WA);             // aliases dead qkv_pre
  float* fpre    = (float*)(ws + WA);             // aliases dead a4f/qkv
  float* fff     = (float*)(ws + WA);             // aliases dead fpre

  // --- weights / DCT matrices to f16 ---
  dctgen<<<144, 256, 0, stream>>>(M16, MT16);
  cvtpad<<<432, 256, 0, stream>>>(wqkv,  wqkv16,  576, 192, 576, 192);
  cvtpad<<<144, 256, 0, stream>>>(wproj, wproj16, 192, 192, 192, 192);
  cvtpad<<<768, 256, 0, stream>>>(win,   win16,  1020, 192, 1024, 192);
  cvtpad<<<384, 256, 0, stream>>>(wout,  wout16,  192, 510, 192, 512);

  // --- attention branch ---
  ln_c<<<288, 256, 0, stream>>>(x, n1w, n1b, f16a);                       // y
  gemm16<<<dim3(3, 3, 384), 128, 0, stream>>>(                            // DCT-H
      M16, f16a, nullptr, f16b, nullptr, 192, 192, 192, 0, HW, HW, 0);
  gemm16<<<dim3(3, 3, 384), 128, 0, stream>>>(                            // DCT-W
      f16b, M16, nullptr, f16a, nullptr, 192, 192, 192, HW, 0, HW, 1);
  gemm16<<<dim3(576, 9, 2), 128, 0, stream>>>(                            // qkv 1x1
      wqkv16, f16a, qkv_pre, nullptr, nullptr, 576, 36864, 192,
      0, 192 * HW, 576 * HW, 0);
  dwconv<<<165888, 256, 0, stream>>>(qkv_pre, wdw, qkvb, 576);            // dw 3x3
  winattn<<<6912, 32, 0, stream>>>(qkvb, temp, f16b);                     // attn
  gemm16<<<dim3(576, 3, 2), 128, 0, stream>>>(                            // proj
      wproj16, f16b, nullptr, f16a, bproj, 192, 36864, 192,
      0, 192 * HW, 192 * HW, 0);
  gemm16<<<dim3(3, 3, 384), 128, 0, stream>>>(                            // IDCT-W
      f16a, M16, nullptr, f16b, nullptr, 192, 192, 192, HW, 0, HW, 0);
  gemm16<<<dim3(3, 3, 384), 128, 0, stream>>>(                            // IDCT-H
      MT16, f16b, a4f, nullptr, nullptr, 192, 192, 192, 0, HW, HW, 0);
  addf<<<55296, 256, 0, stream>>>(x, a4f, out, NEL);                      // x1

  // --- FFN branch ---
  ln_c<<<288, 256, 0, stream>>>(out, n2w, n2b, f16a);                     // z
  gemm16<<<dim3(576, 16, 2), 128, 0, stream>>>(                           // ffn-in
      win16, f16a, fpre, nullptr, nullptr, 1020, 36864, 192,
      0, 192 * HW, 1020 * HW, 0);
  dwgelu<<<147456, 256, 0, stream>>>(fpre, wdwf, gbuf);                   // dw+gelu*gate
  gemm16<<<dim3(576, 3, 2), 128, 0, stream>>>(                            // ffn-out
      wout16, gbuf, fff, nullptr, nullptr, 192, 36864, 512,
      0, 512 * HW, 192 * HW, 0);
  addf<<<55296, 256, 0, stream>>>(out, fff, out, NEL);                    // + residual
}
